// _SelfAttention3D_62182536512321
// MI455X (gfx1250) — matrix operand order repaired and checked by reading, not compiled, not run on hardware
//
#include <hip/hip_runtime.h>

typedef __attribute__((ext_vector_type(16))) _Float16 v16h;
typedef __attribute__((ext_vector_type(8)))  _Float16 v8h;
typedef __attribute__((ext_vector_type(2)))  __fp16   v2hf;   // cvt_pkrtz result type
typedef __attribute__((ext_vector_type(8)))  float    v8f;
typedef int i32x4 __attribute__((vector_size(16)));            // async-LDS builtin elem type

static constexpr int NB = 2;      // batch
static constexpr int C_ = 64;     // channels
static constexpr int CH = 32;     // q/k head dim
static constexpr int NN = 8192;   // spatial voxels (8*32*32)

#if defined(__has_builtin)
#if __has_builtin(__builtin_amdgcn_global_load_async_to_lds_b128)
#define HAVE_ASYNC_LDS 1
#endif
#if __has_builtin(__builtin_amdgcn_permlanex16)
#define HAVE_PERMX16 1
#endif
#endif

// xor-16 lane shuffle (wave32): v_permlanex16_b32 with identity selects.
__device__ __forceinline__ unsigned xor16u(unsigned v) {
#ifdef HAVE_PERMX16
  return __builtin_amdgcn_permlanex16(v, v, 0x76543210u, 0xFEDCBA98u, false, false);
#else
  return (unsigned)__shfl_xor((int)v, 16, 32);
#endif
}
__device__ __forceinline__ float xor16f(float v) {
  return __builtin_bit_cast(float, xor16u(__builtin_bit_cast(unsigned, v)));
}

// 16B global -> LDS copy. Async (ASYNCcnt) when the CDNA5 builtin exists.
__device__ __forceinline__ void gld_async_b128(const void* g, void* l) {
#ifdef HAVE_ASYNC_LDS
  __builtin_amdgcn_global_load_async_to_lds_b128(
      (__attribute__((address_space(1))) i32x4*)(uintptr_t)g,
      (__attribute__((address_space(3))) i32x4*)(unsigned)(uintptr_t)l,
      0, 0);
#else
  *(v8h*)l = *(const v8h*)g;
#endif
}
__device__ __forceinline__ void wait_async() {
#ifdef HAVE_ASYNC_LDS
#if __has_builtin(__builtin_amdgcn_s_wait_asynccnt)
  __builtin_amdgcn_s_wait_asynccnt(0);
#else
  asm volatile("s_wait_asynccnt 0x0" ::: "memory");
#endif
#endif
}

// ---------------------------------------------------------------------------
// Kernel 1: QKV projection (1x1 conv == per-voxel linear). x: f32 [B][C][N].
// Writes f16:  q16 channel-major [B][CH][N], k16 key-major [B][N][CH],
//              v16 channel-major [B][C][N].
// ---------------------------------------------------------------------------
__global__ __launch_bounds__(256) void qkv_proj_k(
    const float* __restrict__ x,
    const float* __restrict__ wq, const float* __restrict__ bq,
    const float* __restrict__ wk, const float* __restrict__ bk,
    const float* __restrict__ wv, const float* __restrict__ bv,
    _Float16* __restrict__ q16, _Float16* __restrict__ k16,
    _Float16* __restrict__ v16)
{
  __shared__ float sWq[CH * C_], sWk[CH * C_], sWv[C_ * C_];
  __shared__ float sBq[CH], sBk[CH], sBv[C_];
  const int t = threadIdx.x;
  for (int i = t; i < CH * C_; i += 256) { sWq[i] = wq[i]; sWk[i] = wk[i]; }
  for (int i = t; i < C_ * C_; i += 256) sWv[i] = wv[i];
  if (t < CH) { sBq[t] = bq[t]; sBk[t] = bk[t]; }
  if (t < C_) sBv[t] = bv[t];
  __syncthreads();

  const int gid = blockIdx.x * 256 + t;     // over B*N
  const int b = gid / NN, n = gid % NN;

  float xv[C_];
  const float* xb = x + (size_t)b * C_ * NN + n;
#pragma unroll
  for (int c = 0; c < C_; ++c) xv[c] = xb[(size_t)c * NN];

  for (int o = 0; o < CH; ++o) {
    float aq = sBq[o], ak = sBk[o];
#pragma unroll
    for (int c = 0; c < C_; ++c) {
      aq += sWq[o * C_ + c] * xv[c];
      ak += sWk[o * C_ + c] * xv[c];
    }
    q16[((size_t)b * CH + o) * NN + n]   = (_Float16)aq;
    k16[((size_t)(b * NN + n)) * CH + o] = (_Float16)ak;
  }
  for (int o = 0; o < C_; ++o) {
    float av = sBv[o];
#pragma unroll
    for (int c = 0; c < C_; ++c) av += sWv[o * C_ + c] * xv[c];
    v16[((size_t)b * C_ + o) * NN + n] = (_Float16)av;
  }
}

// ---------------------------------------------------------------------------
// Kernel 2: flash attention. 256 thr = 8 waves; each wave owns a 16-query
// tile; block streams all 8192 keys in 32-key stages through double-buffered
// LDS (async global->LDS). Per stage per wave: 2x WMMA (S^T = K*Q^T),
// in-register softmax + P-fragment rebuild (cvt_pkrtz + permlanex16),
// 4x WMMA (O^T += V^T * P^T).
// ---------------------------------------------------------------------------
__global__ __launch_bounds__(256) void flash_attn_k(
    const _Float16* __restrict__ q16, const _Float16* __restrict__ k16,
    const _Float16* __restrict__ v16, _Float16* __restrict__ attn16)
{
  __shared__ __align__(16) _Float16 sK[2][32 * CH];  // [key][c]
  __shared__ __align__(16) _Float16 sV[2][C_ * 32];  // [c][key]

  const int t    = threadIdx.x;
  const int lane = t & 31;
  const int wave = t >> 5;                 // 0..7
  const int bpb  = NN / 128;               // blocks per batch = 64
  const int b    = blockIdx.x / bpb;
  const int nbase = (blockIdx.x % bpb) * 128 + wave * 16;
  const bool lo16 = (lane < 16);
  const int  nloc = lane & 15;

  const _Float16* qb = q16 + (size_t)b * CH * NN;
  const _Float16* kb = k16 + (size_t)b * NN * CH;
  const _Float16* vb = v16 + (size_t)b * C_ * NN;

  // ---- Q B-fragment (loaded once): B[c,n] tile [32ch x 16q] ----
  union BF { v16h h; _Float16 e[16]; unsigned u[8]; };
  BF Bq;
#pragma unroll
  for (int j = 0; j < 8; ++j) {
    const int c = 2 * j + (j >= 4 ? 8 : 0) + (lo16 ? 0 : 8);
    Bq.e[2 * j]     = qb[(size_t)c * NN + nbase + nloc];
    Bq.e[2 * j + 1] = qb[(size_t)(c + 1) * NN + nbase + nloc];
  }

  const v8f vzero = {};
  v8f acc[4] = {vzero, vzero, vzero, vzero};   // O^T tiles, c in 4 tiles of 16
  float mrun = -1e30f, lrun = 0.0f;
  const float sl2e = 0.17677669529663688f * 1.4426950408889634f; // scale*log2e

  auto load_stage = [&](int stage, int buf) {
    const int m0 = stage * 32;
    if (t < 128) {                       // K: 32 rows x 64B
      const int r = t >> 2, seg = t & 3;
      gld_async_b128((const _Float16*)(kb + (size_t)(m0 + r) * CH) + seg * 8,
                     &sK[buf][r * CH + seg * 8]);
    }
    {                                    // V: 64 rows x 64B (channel-major)
      const int c = t >> 2, seg = t & 3;
      gld_async_b128((const _Float16*)(vb + (size_t)c * NN + m0) + seg * 8,
                     &sV[buf][c * 32 + seg * 8]);
    }
  };

  load_stage(0, 0);
  wait_async();
  __syncthreads();

  for (int stage = 0; stage < NN / 32; ++stage) {
    const int buf = stage & 1;
    if (stage + 1 < NN / 32) load_stage(stage + 1, buf ^ 1);

    // ---- S^T tiles: A = K rows [16keys x 32ch] ----
    union AF { v16h h; v8h p[2]; };
    const _Float16* Kb = &sK[buf][0];
    const int krow = lane & 15;
    const int coff = lo16 ? 0 : 8;
    AF Ak0, Ak1;
    Ak0.p[0] = *(const v8h*)(Kb + krow * CH + coff);
    Ak0.p[1] = *(const v8h*)(Kb + krow * CH + coff + 16);
    Ak1.p[0] = *(const v8h*)(Kb + (16 + krow) * CH + coff);
    Ak1.p[1] = *(const v8h*)(Kb + (16 + krow) * CH + coff + 16);

    v8f S0 = __builtin_amdgcn_wmma_f32_16x16x32_f16(
        false, Ak0.h, false, Bq.h, (short)0, vzero, false, false);
    v8f S1 = __builtin_amdgcn_wmma_f32_16x16x32_f16(
        false, Ak1.h, false, Bq.h, (short)0, vzero, false, false);

    // ---- online softmax (lane pair l / l+16 share one query) ----
    float p0[8], p1[8], tmax = -1e30f;
#pragma unroll
    for (int j = 0; j < 8; ++j) {
      p0[j] = S0[j] * sl2e; p1[j] = S1[j] * sl2e;
      tmax = fmaxf(tmax, fmaxf(p0[j], p1[j]));
    }
    tmax = fmaxf(tmax, xor16f(tmax));
    const float mnew = fmaxf(mrun, tmax);
    const float corr = __builtin_amdgcn_exp2f(mrun - mnew);
    float ssum = 0.0f;
#pragma unroll
    for (int j = 0; j < 8; ++j) {
      p0[j] = __builtin_amdgcn_exp2f(p0[j] - mnew);
      p1[j] = __builtin_amdgcn_exp2f(p1[j] - mnew);
      ssum += p0[j] + p1[j];
    }
    ssum += xor16f(ssum);
    lrun = lrun * corr + ssum;
    mrun = mnew;
#pragma unroll
    for (int cc = 0; cc < 4; ++cc)
#pragma unroll
      for (int j = 0; j < 8; ++j) acc[cc][j] *= corr;

    // ---- rebuild P^T as a B-fragment, in registers ----
    unsigned pk0[4], pk1[4];
#pragma unroll
    for (int j = 0; j < 4; ++j) {
      union { v2hf h; unsigned u; } cv;
      cv.h = __builtin_amdgcn_cvt_pkrtz(p0[2 * j], p0[2 * j + 1]); pk0[j] = cv.u;
      cv.h = __builtin_amdgcn_cvt_pkrtz(p1[2 * j], p1[2 * j + 1]); pk1[j] = cv.u;
    }
    BF Bp;
#pragma unroll
    for (int j = 0; j < 4; ++j) {
      const unsigned x0 = xor16u(pk0[j]);
      const unsigned x1 = xor16u(pk1[j]);
      Bp.u[j]     = lo16 ? pk0[j] : x1;   // keys 2j,2j+1   (+16 for hi lanes)
      Bp.u[4 + j] = lo16 ? x0 : pk1[j];   // keys 8+2j,...  (+16 for hi lanes)
    }

    // ---- O^T += V^T * P^T : A = V^T [16ch x 32keys], 4 channel tiles ----
    const _Float16* Vb = &sV[buf][0];
#pragma unroll
    for (int cc = 0; cc < 4; ++cc) {
      AF Av;
      const int crow = cc * 16 + (lane & 15);
      Av.p[0] = *(const v8h*)(Vb + crow * 32 + (lo16 ? 0 : 16));
      Av.p[1] = *(const v8h*)(Vb + crow * 32 + (lo16 ? 0 : 16) + 8);
      acc[cc] = __builtin_amdgcn_wmma_f32_16x16x32_f16(
          false, Av.h, false, Bp.h, (short)0, acc[cc], false, false);
    }
    wait_async();
    __syncthreads();
  }

  // ---- epilogue: normalize and store attn (channel-major f16) ----
  const float inv = 1.0f / lrun;
#pragma unroll
  for (int cc = 0; cc < 4; ++cc)
#pragma unroll
    for (int j = 0; j < 8; ++j) {
      const int c = cc * 16 + j + (lo16 ? 0 : 8);
      attn16[((size_t)b * C_ + c) * NN + nbase + nloc] =
          (_Float16)(acc[cc][j] * inv);
    }
}

// ---------------------------------------------------------------------------
// Kernel 3: output projection, f32 out = wo @ attn + bo.
// ---------------------------------------------------------------------------
__global__ __launch_bounds__(256) void out_proj_k(
    const _Float16* __restrict__ attn16, const float* __restrict__ wo,
    const float* __restrict__ bo, float* __restrict__ out)
{
  __shared__ float sW[C_ * C_];
  __shared__ float sB[C_];
  const int t = threadIdx.x;
  for (int i = t; i < C_ * C_; i += 256) sW[i] = wo[i];
  if (t < C_) sB[t] = bo[t];
  __syncthreads();

  const int gid = blockIdx.x * 256 + t;
  const int b = gid / NN, n = gid % NN;
  float av[C_];
  const _Float16* ab = attn16 + (size_t)b * C_ * NN + n;
#pragma unroll
  for (int c = 0; c < C_; ++c) av[c] = (float)ab[(size_t)c * NN];
  for (int o = 0; o < C_; ++o) {
    float s = sB[o];
#pragma unroll
    for (int c = 0; c < C_; ++c) s += sW[o * C_ + c] * av[c];
    out[((size_t)b * C_ + o) * NN + n] = s;
  }
}

extern "C" void kernel_launch(void* const* d_in, const int* in_sizes, int n_in,
                              void* d_out, int out_size, void* d_ws, size_t ws_size,
                              hipStream_t stream)
{
  const float* x  = (const float*)d_in[0];
  const float* wq = (const float*)d_in[1];
  const float* bq = (const float*)d_in[2];
  const float* wk = (const float*)d_in[3];
  const float* bk = (const float*)d_in[4];
  const float* wv = (const float*)d_in[5];
  const float* bv = (const float*)d_in[6];
  const float* wo = (const float*)d_in[7];
  const float* bo = (const float*)d_in[8];
  float* out = (float*)d_out;

  _Float16* q16 = (_Float16*)d_ws;                       // [B][CH][N]  1 MB
  _Float16* k16 = q16 + (size_t)NB * CH * NN;            // [B][N][CH]  1 MB
  _Float16* v16 = k16 + (size_t)NB * NN * CH;            // [B][C][N]   2 MB
  _Float16* a16 = v16 + (size_t)NB * C_ * NN;            // [B][C][N]   2 MB

  qkv_proj_k<<<(NB * NN) / 256, 256, 0, stream>>>(x, wq, bq, wk, bk, wv, bv,
                                                  q16, k16, v16);
  flash_attn_k<<<(NB * NN) / 128, 256, 0, stream>>>(q16, k16, v16, a16);
  out_proj_k<<<(NB * NN) / 256, 256, 0, stream>>>(a16, wo, bo, out);
}
